// InterpretableMultiHeadAttention_1803886264638
// MI455X (gfx1250) — matrix operand order repaired and checked by reading, not compiled, not run on hardware
//
#include <hip/hip_runtime.h>
#include <hip/hip_bf16.h>
#include <hip/hip_fp16.h>

// ---- problem constants -----------------------------------------------------
#define DMODEL  1024
#define NHEADS  8
#define HEADDIM 128
#define BATCH   4
#define SEQ     2048
#define BT      (BATCH * SEQ)   // 8192 total rows

typedef __attribute__((ext_vector_type(16))) _Float16 v16h;  // 8 VGPRs
typedef __attribute__((ext_vector_type(8)))  _Float16 v8h;   // 4 VGPRs
typedef __attribute__((ext_vector_type(8)))  float    v8f;   // 8 VGPRs

// ---- WMMA fragment helpers (CDNA5 16x16x32 f16 layouts, ISA 7.12.2) --------
// A (16x32, f16): lane L<16 holds row L, K=k0+0..7 (elts0-7) and K=k0+16..23
// (elts8-15); lane L+16 holds row L, K=k0+8..15 / k0+24..31.
__device__ __forceinline__ v16h load_a_f16(const _Float16* __restrict__ A,
                                           int lda, int r0, int k0, int lane) {
  const int r = lane & 15, ko = (lane >> 4) * 8;
  const _Float16* p = A + (size_t)(r0 + r) * lda + k0 + ko;
  v8h lo = *(const v8h*)p;
  v8h hi = *(const v8h*)(p + 16);
  return __builtin_shufflevector(lo, hi, 0, 1, 2, 3, 4, 5, 6, 7,
                                 8, 9, 10, 11, 12, 13, 14, 15);
}

// Same A fragment but sourced from f32 (converted inline) — used for alpha.
__device__ __forceinline__ v16h load_a_f32cvt(const float* __restrict__ A,
                                              int lda, int r0, int k0, int lane) {
  const int r = lane & 15, ko = (lane >> 4) * 8;
  const float* p = A + (size_t)(r0 + r) * lda + k0 + ko;
  v8f lo = *(const v8f*)p;
  v8f hi = *(const v8f*)(p + 16);
  v16h out;
#pragma unroll
  for (int i = 0; i < 8; ++i) {
    out[i]     = (_Float16)lo[i];
    out[i + 8] = (_Float16)hi[i];
  }
  return out;
}

// B (32x16, f16) where the operand is stored row-major as [N,K] (Linear W,
// K rows, VT rows): lane L<16 holds column L, K=k0+0..7 (elts0-7) and
// K=k0+16..23 (elts8-15); lane L+16 holds column L, K=k0+8..15 / k0+24..31.
__device__ __forceinline__ v16h load_b_f16(const _Float16* __restrict__ Bw,
                                           int ldb, int c0, int k0, int lane) {
  const int c = lane & 15, kk = (lane >> 4) * 8;
  const _Float16* p = Bw + (size_t)(c0 + c) * ldb + k0 + kk;
  v8h lo = *(const v8h*)p;
  v8h hi = *(const v8h*)(p + 16);
  return __builtin_shufflevector(lo, hi, 0, 1, 2, 3, 4, 5, 6, 7,
                                 8, 9, 10, 11, 12, 13, 14, 15);
}

__device__ __forceinline__ v8f wmma_f16(v16h a, v16h b, v8f c) {
  return __builtin_amdgcn_wmma_f32_16x16x32_f16(false, a, false, b, (short)0, c,
                                                false, false);
}

// ---- f32 -> f16 conversion -------------------------------------------------
__global__ void cvt_f32_to_f16_kernel(const float* __restrict__ s,
                                      _Float16* __restrict__ d, int n) {
  int i = blockIdx.x * blockDim.x + threadIdx.x;
  if (i < n) d[i] = (_Float16)s[i];
}

// ---- generic WMMA GEMM: C = A * B^T (+bias) --------------------------------
// A: [M,K] (f16 or f32->f16), B stored as [N,K] rows (ldb stride),
// SMODE 0: f16 C[m*ldc+n]; 1: f16 transposed C[n*ldct+m]; 2: f32 C[m*ldc+n].
// Workgroup (8 waves) -> 128x64 tile; wave -> 32x32 (2x2 WMMA tiles).
template <int AMODE, int SMODE>
__global__ __launch_bounds__(256) void gemm_wmma_kernel(
    const void* __restrict__ Av, const _Float16* __restrict__ Bw,
    const float* __restrict__ bias, void* __restrict__ Cv, int Kdim, int lda,
    int ldb, int ldc, int ldct, long sA, long sB, long sC) {
  const int tid = threadIdx.x;
  const int wave = tid >> 5, lane = tid & 31;
  const int row0 = blockIdx.y * 128 + (wave >> 1) * 32;
  const int col0 = blockIdx.x * 64 + (wave & 1) * 32;
  const int z = blockIdx.z;
  const _Float16* Bb = Bw + (size_t)z * sB;

  v8f acc[2][2] = {};

  for (int k0 = 0; k0 < Kdim; k0 += 32) {
    if (k0 + 32 < Kdim)  // global_prefetch_b8 for the next K slab of B
      __builtin_prefetch(Bb + (size_t)(col0 + (lane & 15)) * ldb + k0 + 32, 0, 0);
    v16h a0, a1;
    if (AMODE == 0) {
      const _Float16* A = (const _Float16*)Av + (size_t)z * sA;
      a0 = load_a_f16(A, lda, row0, k0, lane);
      a1 = load_a_f16(A, lda, row0 + 16, k0, lane);
    } else {
      const float* A = (const float*)Av + (size_t)z * sA;
      a0 = load_a_f32cvt(A, lda, row0, k0, lane);
      a1 = load_a_f32cvt(A, lda, row0 + 16, k0, lane);
    }
    v16h b0 = load_b_f16(Bb, ldb, col0, k0, lane);
    v16h b1 = load_b_f16(Bb, ldb, col0 + 16, k0, lane);
    acc[0][0] = wmma_f16(a0, b0, acc[0][0]);
    acc[0][1] = wmma_f16(a0, b1, acc[0][1]);
    acc[1][0] = wmma_f16(a1, b0, acc[1][0]);
    acc[1][1] = wmma_f16(a1, b1, acc[1][1]);
  }

  // Epilogue. C tile layout: elt j <-> row (j or j+8 for upper half-wave),
  // col = lane&15.
  const int cn = lane & 15, rh = (lane >> 4) * 8;
#pragma unroll
  for (int ct = 0; ct < 2; ++ct) {
    const int col = col0 + ct * 16 + cn;
    const float bb = bias ? bias[col] : 0.0f;
#pragma unroll
    for (int rt = 0; rt < 2; ++rt) {
      const int rbase = row0 + rt * 16 + rh;
      if (SMODE == 0) {
        _Float16* C = (_Float16*)Cv + (size_t)z * sC;
#pragma unroll
        for (int j = 0; j < 8; ++j)
          C[(size_t)(rbase + j) * ldc + col] = (_Float16)(acc[rt][ct][j] + bb);
      } else if (SMODE == 1) {
        _Float16* C = (_Float16*)Cv + (size_t)z * sC;
        v8h pk;
#pragma unroll
        for (int j = 0; j < 8; ++j) pk[j] = (_Float16)(acc[rt][ct][j] + bb);
        *(v8h*)(C + (size_t)col * ldct + rbase) = pk;  // contiguous along rows
      } else {
        float* C = (float*)Cv + (size_t)z * sC;
#pragma unroll
        for (int j = 0; j < 8; ++j)
          C[(size_t)(rbase + j) * ldc + col] = acc[rt][ct][j] + bb;
      }
    }
  }
}

// ---- attention: head-averaged causal softmax -> alpha_agg [B,T,T] ----------
// One workgroup per (batch, 16-row q tile); wave w == head w (8 waves).
// Pass 1: online (rowmax, sumexp) per head. Pass 2: recompute scores,
// p = exp(s-m)/l/H, head-average via LDS ds_add_f32 in 16x512 chunks.
__global__ __launch_bounds__(256) void attn_alpha_kernel(
    const _Float16* __restrict__ Qh, const _Float16* __restrict__ Kh,
    float* __restrict__ alpha) {
  __shared__ float asmem[16 * 512];  // 32 KB accumulation chunk
  const int tid = threadIdx.x;
  const int h = tid >> 5;     // wave index == head
  const int lane = tid & 31;
  const int b = blockIdx.x / (SEQ / 16);
  const int qt = blockIdx.x % (SEQ / 16);
  const int q0 = qt * 16;
  const float scale = 0.088388347648318447f;  // 1/sqrt(128)

  const _Float16* Qbase = Qh + (size_t)(b * SEQ + q0) * DMODEL + h * HEADDIM;
  const _Float16* Kbase = Kh + (size_t)b * SEQ * DMODEL + h * HEADDIM;

  v16h qf[4];  // 16 rows x Dh=128 (4 K-steps of 32)
#pragma unroll
  for (int ks = 0; ks < 4; ++ks) qf[ks] = load_a_f16(Qbase, DMODEL, 0, ks * 32, lane);

  const int cn = lane & 15;
  const int rh = (lane >> 4) * 8;

  float m[8], l[8];
#pragma unroll
  for (int j = 0; j < 8; ++j) { m[j] = -3.0e38f; l[j] = 0.0f; }

  // ---- pass 1: per-row running max / sumexp over causal key tiles ----
  for (int kt = 0; kt <= qt; ++kt) {
    v8f acc = {};
#pragma unroll
    for (int ks = 0; ks < 4; ++ks)
      acc = wmma_f16(qf[ks], load_b_f16(Kbase, DMODEL, kt * 16, ks * 32, lane), acc);
    const int kcol = kt * 16 + cn;
#pragma unroll
    for (int j = 0; j < 8; ++j) {
      const int qrow = q0 + rh + j;
      float s = (kcol <= qrow) ? acc[j] * scale : -3.0e38f;
      float tmax = s;
#pragma unroll
      for (int d = 1; d < 16; d <<= 1) tmax = fmaxf(tmax, __shfl_xor(tmax, d, 32));
      const float mn = fmaxf(m[j], tmax);
      float ts = __expf(s - mn);
#pragma unroll
      for (int d = 1; d < 16; d <<= 1) ts += __shfl_xor(ts, d, 32);
      l[j] = l[j] * __expf(m[j] - mn) + ts;
      m[j] = mn;
    }
  }

  float w[8];
#pragma unroll
  for (int j = 0; j < 8; ++j) w[j] = 1.0f / (l[j] * (float)NHEADS);

  // ---- pass 2: recompute, normalize, head-average, emit alpha ----
  for (int chunk = 0; chunk < SEQ / 512; ++chunk) {
    for (int i = tid; i < 16 * 512; i += 256) asmem[i] = 0.0f;
    __syncthreads();

    const int ktend = min(qt, chunk * 32 + 31);
    for (int kt = chunk * 32; kt <= ktend; ++kt) {
      v8f acc = {};
#pragma unroll
      for (int ks = 0; ks < 4; ++ks)
        acc = wmma_f16(qf[ks], load_b_f16(Kbase, DMODEL, kt * 16, ks * 32, lane), acc);
      const int kcol = kt * 16 + cn;
      const int klocal = (kt - chunk * 32) * 16 + cn;
#pragma unroll
      for (int j = 0; j < 8; ++j) {
        const int qrow = q0 + rh + j;
        const float s = (kcol <= qrow) ? acc[j] * scale : -3.0e38f;
        atomicAdd(&asmem[(rh + j) * 512 + klocal], __expf(s - m[j]) * w[j]);
      }
    }
    __syncthreads();

    float* dst = alpha + (size_t)(b * SEQ + q0) * SEQ + chunk * 512;
    const int r = tid >> 4;
    const int cb = (tid & 15) * 32;
#pragma unroll
    for (int c = 0; c < 32; c += 4)
      *(float4*)(dst + (size_t)r * SEQ + cb + c) =
          *(const float4*)(asmem + r * 512 + cb + c);
    __syncthreads();
  }
}

// ---- host-side launcher ----------------------------------------------------
extern "C" void kernel_launch(void* const* d_in, const int* in_sizes, int n_in,
                              void* d_out, int out_size, void* d_ws,
                              size_t ws_size, hipStream_t stream) {
  (void)in_sizes; (void)n_in; (void)out_size; (void)ws_size;
  const float* x  = (const float*)d_in[0];
  // d_in[1] = mask (causal, recomputed analytically)
  const float* Wq = (const float*)d_in[2];
  const float* bq = (const float*)d_in[3];
  const float* Wk = (const float*)d_in[4];
  const float* bk = (const float*)d_in[5];
  const float* Wv = (const float*)d_in[6];
  const float* bv = (const float*)d_in[7];
  const float* Wo = (const float*)d_in[8];
  const float* bo = (const float*)d_in[9];

  float* out   = (float*)d_out;                  // [B,T,D]
  float* alpha = out + (size_t)BT * DMODEL;      // [B,T,T] (output #2)

  // workspace layout (f16), ~88 MB total
  _Float16* xh   = (_Float16*)d_ws;
  _Float16* wqh  = xh   + (size_t)BT * DMODEL;
  _Float16* wkh  = wqh  + (size_t)DMODEL * DMODEL;
  _Float16* wvh  = wkh  + (size_t)DMODEL * DMODEL;
  _Float16* woh  = wvh  + (size_t)DMODEL * DMODEL;
  _Float16* Qhp  = woh  + (size_t)DMODEL * DMODEL;
  _Float16* Khp  = Qhp  + (size_t)BT * DMODEL;
  _Float16* VTh  = Khp  + (size_t)BT * DMODEL;   // [D][BT] (V transposed)
  _Float16* ctxh = VTh  + (size_t)BT * DMODEL;   // [BT][D]

  const int nx = BT * DMODEL, nw = DMODEL * DMODEL;
  cvt_f32_to_f16_kernel<<<(nx + 255) / 256, 256, 0, stream>>>(x, xh, nx);
  cvt_f32_to_f16_kernel<<<(nw + 255) / 256, 256, 0, stream>>>(Wq, wqh, nw);
  cvt_f32_to_f16_kernel<<<(nw + 255) / 256, 256, 0, stream>>>(Wk, wkh, nw);
  cvt_f32_to_f16_kernel<<<(nw + 255) / 256, 256, 0, stream>>>(Wv, wvh, nw);
  cvt_f32_to_f16_kernel<<<(nw + 255) / 256, 256, 0, stream>>>(Wo, woh, nw);

  const dim3 blk(256);
  const dim3 gproj(DMODEL / 64, BT / 128, 1);
  // Q = xh @ Wq^T + bq   (f16 out)
  gemm_wmma_kernel<0, 0><<<gproj, blk, 0, stream>>>(
      xh, wqh, bq, Qhp, DMODEL, DMODEL, DMODEL, DMODEL, 0, 0, 0, 0);
  // K = xh @ Wk^T + bk   (f16 out)
  gemm_wmma_kernel<0, 0><<<gproj, blk, 0, stream>>>(
      xh, wkh, bk, Khp, DMODEL, DMODEL, DMODEL, DMODEL, 0, 0, 0, 0);
  // VT = (xh @ Wv^T + bv)^T  (f16 transposed store, VT[d][m])
  gemm_wmma_kernel<0, 1><<<gproj, blk, 0, stream>>>(
      xh, wvh, bv, VTh, DMODEL, DMODEL, DMODEL, 0, BT, 0, 0, 0);

  // alpha_agg (head-averaged causal softmax) -> d_out region 2
  attn_alpha_kernel<<<BATCH * (SEQ / 16), blk, 0, stream>>>(Qhp, Khp, alpha);

  // ctx = alpha @ V  (batched: A f32 from d_out, B = VT rows, f16 out)
  const dim3 gctx(DMODEL / 64, SEQ / 128, BATCH);
  gemm_wmma_kernel<1, 0><<<gctx, blk, 0, stream>>>(
      alpha, VTh, nullptr, ctxh, SEQ, SEQ, BT, DMODEL, 0,
      (long)SEQ * SEQ, (long)SEQ, (long)SEQ * DMODEL);

  // out = ctx @ Wo^T + bo  (f32 out -> d_out region 1)
  gemm_wmma_kernel<0, 2><<<gproj, blk, 0, stream>>>(
      ctxh, woh, bo, out, DMODEL, DMODEL, DMODEL, DMODEL, 0, 0, 0, 0);
}
